// EGNN_predictor_40604620816578
// MI455X (gfx1250) — matrix operand order repaired and checked by reading, not compiled, not run on hardware
//
#include <hip/hip_runtime.h>
#include <hip/hip_bf16.h>
#include <math.h>
#include <stdint.h>

// ---------------------------------------------------------------------------
// EGNN forward for MI455X (gfx1250, wave32, WMMA 16x16x32 f16).
// B=64 graphs x N=64 nodes, H=128, L=4, fully-connected edges.
// Heavy GEMMs on v_wmma_f32_16x16x32_f16; weights pre-transposed to f16 so
// B-fragments are contiguous LDS loads; weight tiles staged to LDS with
// global_load_async_to_lds_b128 (ASYNCcnt) overlapping VALU work.
// ---------------------------------------------------------------------------

typedef __attribute__((ext_vector_type(16))) _Float16 v16h;
typedef __attribute__((ext_vector_type(8)))  _Float16 v8h;
typedef __attribute__((ext_vector_type(8)))  float    v8f;

#define WMMA_F16(A,B,C) \
  __builtin_amdgcn_wmma_f32_16x16x32_f16(false,(A),false,(B),(short)0,(C),false,false)

#define HD        128      // hidden dim
#define NPG       64       // nodes per graph
#define BN        4096     // total nodes
#define LAYERS    4
#define SA_LD     136      // f16 LDS row stride (halves), 16B-aligned rows
#define SW_LD     136      // f16 LDS weight-T row stride (halves)
#define WPL       114688   // f16 weight halves per layer
#define OF_EW1A   0        // all weight regions stored TRANSPOSED: WT[n][k]
#define OF_EW1B   16384
#define OF_EW2    32768
#define OF_CW1    49152
#define OF_NW1    65536    // WT [128][256]
#define OF_NW2    98304
#define EW1_LSTR  33024    // 258*128 f32 per layer

__device__ __forceinline__ float silu_f(float x) {
  return x / (1.0f + __expf(-x));
}

// A-fragment (16x32 f16, ISA 7.12.2): lane l<16 row M=m0+l holds K=k0+{0..7,16..23},
// lanes 16..31 hold K=k0+{8..15,24..31}. Rows 16B aligned -> two ds_load_b128.
__device__ __forceinline__ v16h frag_a(const _Float16* A, int lda, int m0, int k0) {
  int lane = threadIdx.x & 31;
  int l = lane & 15, hb = lane >> 4;
  const _Float16* p = A + (m0 + l) * lda + k0 + hb * 8;
  v8h lo = *(const v8h*)(p);
  v8h hi = *(const v8h*)(p + 16);
  return __builtin_shufflevector(lo, hi, 0,1,2,3,4,5,6,7,8,9,10,11,12,13,14,15);
}

// B-fragment from TRANSPOSED weights WT[n][k] (stride ldb halves):
// lane l<16 -> column N=n0+l needs K=k0+hb*8+{0..7,16..23} -> two 16B loads.
__device__ __forceinline__ v16h frag_bT(const _Float16* WT, int ldb, int k0, int n0) {
  int lane = threadIdx.x & 31;
  int l = lane & 15, hb = lane >> 4;
  const _Float16* p = WT + (n0 + l) * ldb + k0 + hb * 8;
  v8h lo = *(const v8h*)(p);
  v8h hi = *(const v8h*)(p + 16);
  return __builtin_shufflevector(lo, hi, 0,1,2,3,4,5,6,7,8,9,10,11,12,13,14,15);
}

// ---------------------------------------------------------------------------
// Async LDS staging (CDNA5): global_load_async_to_lds_b128, GV addressing.
// Copies a 128x128-half WT tile (global row stride gstride halves) into LDS
// with SW_LD-padded rows. 2048 x 16B chunks across 256 threads.
// ---------------------------------------------------------------------------
__device__ __forceinline__ void stage_wT_async(_Float16* sW, const _Float16* WT,
                                               int gstride, int tid) {
  for (int c = tid; c < 2048; c += 256) {
    int row = c >> 4, ch = c & 15;
    uint32_t laddr = (uint32_t)(uintptr_t)(sW + row * SW_LD + ch * 8);
    uint64_t gaddr = (uint64_t)(uintptr_t)(WT + (size_t)row * gstride + ch * 8);
    asm volatile("global_load_async_to_lds_b128 %0, %1, off"
                 :: "v"(laddr), "v"(gaddr) : "memory");
  }
}
__device__ __forceinline__ void async_wait_all() {
  asm volatile("s_wait_asynccnt 0x0" ::: "memory");
}

// ---------------------------------------------------------------------------
// Weight conversion + transpose: per-layer f16 [ew1a|ew1b|ew2|cw1|nw1|nw2],
// each region stored column-major (WT[n][k]) for direct B-fragment loads.
// ---------------------------------------------------------------------------
__global__ void k_wcvt(const float* __restrict__ ew1, const float* __restrict__ ew2,
                       const float* __restrict__ nw1, const float* __restrict__ nw2,
                       const float* __restrict__ cw1, _Float16* __restrict__ wf16) {
  int idx = blockIdx.x * blockDim.x + threadIdx.x;
  if (idx >= LAYERS * WPL) return;
  int l = idx / WPL;
  int r = idx - l * WPL;
  float v;
  if (r < 16384) {                      // ew1 rows 0..127 (h_i part)
    int n = r >> 7, k = r & 127;
    v = ew1[l * EW1_LSTR + k * HD + n];
  } else if (r < 32768) {               // ew1 rows 128..255 (h_j part)
    int rr = r - 16384; int n = rr >> 7, k = rr & 127;
    v = ew1[l * EW1_LSTR + (128 + k) * HD + n];
  } else if (r < 49152) {
    int rr = r - 32768; int n = rr >> 7, k = rr & 127;
    v = ew2[l * 16384 + k * HD + n];
  } else if (r < 65536) {
    int rr = r - 49152; int n = rr >> 7, k = rr & 127;
    v = cw1[l * 16384 + k * HD + n];
  } else if (r < 98304) {               // nw1: K=256 -> WT [128][256]
    int rr = r - 65536; int n = rr >> 8, k = rr & 255;
    v = nw1[l * 32768 + k * HD + n];
  } else {
    int rr = r - 98304; int n = rr >> 7, k = rr & 127;
    v = nw2[l * 16384 + k * HD + n];
  }
  wf16[idx] = (_Float16)v;
}

// ---------------------------------------------------------------------------
// Prep: masked coords (x0 and x working copy) + embedding h = (feat*nm)@emb_w+b.
// ---------------------------------------------------------------------------
__global__ void k_prep(const float* __restrict__ xh, const float* __restrict__ nmask,
                       const float* __restrict__ emb_w, const float* __restrict__ emb_b,
                       float* __restrict__ x0, float* __restrict__ xA,
                       float* __restrict__ h) {
  int n = blockIdx.x;
  int t = threadIdx.x;
  float m = nmask[n];
  const float* src = xh + (size_t)n * 11;
  if (t < 3) { float v = src[t] * m; x0[n * 3 + t] = v; xA[n * 3 + t] = v; }
  float acc = emb_b[t];
#pragma unroll
  for (int c = 0; c < 8; ++c) acc += src[3 + c] * m * emb_w[c * HD + t];
  h[(size_t)n * HD + t] = acc;
}

// ---------------------------------------------------------------------------
// Hi/Hj precompute: Hi = h@ew1[0:128], Hj = h@ew1[128:256].  64 blocks x 256.
// ---------------------------------------------------------------------------
__launch_bounds__(256)
__global__ void k_hih(const float* __restrict__ h, const _Float16* __restrict__ wl,
                      float* __restrict__ Hi, float* __restrict__ Hj) {
  __shared__ __align__(16) _Float16 sA[NPG * SA_LD];
  __shared__ __align__(16) _Float16 sW[HD * SW_LD];
  int g = blockIdx.x, tid = threadIdx.x, base = g * NPG;
  int w = tid >> 5, lane = tid & 31, l = lane & 15, hb = lane >> 4;
  int m0 = (w >> 1) * 16, nb = (w & 1) * 64;

  for (int idx = tid; idx < NPG * HD; idx += 256) {
    int j = idx >> 7, k = idx & 127;
    sA[j * SA_LD + k] = (_Float16)h[(size_t)(base + j) * HD + k];
  }
  for (int pass = 0; pass < 2; ++pass) {
    const _Float16* WT = wl + (pass ? OF_EW1B : OF_EW1A);
    __syncthreads();                       // guards sA (pass0) / sW reuse (pass1)
    stage_wT_async(sW, WT, HD, tid);
    async_wait_all();
    __syncthreads();
    v8f acc[4] = {};
    for (int kk = 0; kk < HD; kk += 32) {
      v16h a = frag_a(sA, SA_LD, m0, kk);
#pragma unroll
      for (int t = 0; t < 4; ++t) {
        v16h bf = frag_bT(sW, SW_LD, kk, nb + t * 16);
        acc[t] = WMMA_F16(a, bf, acc[t]);
      }
    }
    float* dst = pass ? Hj : Hi;
#pragma unroll
    for (int t = 0; t < 4; ++t) {
      int n = nb + t * 16 + l;
#pragma unroll
      for (int r = 0; r < 8; ++r) {
        int m = m0 + hb * 8 + r;
        dst[(size_t)(base + m) * HD + n] = acc[t][r];
      }
    }
  }
}

// ---------------------------------------------------------------------------
// Edge kernel: one block per (graph b, source node i); j = 0..63 edges.
// m1 = silu(Hi[i]+Hj[j]+radial*wr+ea*we+eb1)      (rank-structured first GEMM)
// m  = silu(m1@ew2+eb2)*em     p = silu(m@cw1+cb1)   s = p@cw2
// agg[i] = sum_j m ; x_next[i] = x[i] + sum_j coord_diff*s*em   (exclusive rows)
// ---------------------------------------------------------------------------
__launch_bounds__(256)
__global__ void k_edge(const float* __restrict__ xc, const float* __restrict__ x0,
                       const float* __restrict__ Hi, const float* __restrict__ Hj,
                       const _Float16* __restrict__ wf16, int layer,
                       const float* __restrict__ ew1, const float* __restrict__ eb1,
                       const float* __restrict__ eb2, const float* __restrict__ cb1,
                       const float* __restrict__ cw2, const float* __restrict__ emask,
                       float* __restrict__ xn, float* __restrict__ agg) {
  __shared__ __align__(16) _Float16 sM[NPG * SA_LD];       // m1 then m (f16 A-tile)
  __shared__ __align__(16) char     sWraw[HD * SW_LD * 2]; // ew2 -> cw1 -> p(f32)
  __shared__ float sCD[NPG * 3], sRad[NPG], sEA[NPG], sEM[NPG], sS[NPG];
  _Float16* sW = (_Float16*)sWraw;
  float*    sP = (float*)sWraw;                            // 64*128*4 <= sizeof(sWraw)

  int tid = threadIdx.x;
  int nodei = blockIdx.x;
  int gbase = (nodei >> 6) << 6;

  const _Float16* wl = wf16 + layer * WPL;
  stage_wT_async(sW, wl + OF_EW2, HD, tid);      // overlap DMA with geometry/m1

  if (tid < NPG) {
    int nodej = gbase + tid;
    float dx = xc[nodei * 3 + 0] - xc[nodej * 3 + 0];
    float dy = xc[nodei * 3 + 1] - xc[nodej * 3 + 1];
    float dz = xc[nodei * 3 + 2] - xc[nodej * 3 + 2];
    float rad = dx * dx + dy * dy + dz * dz;
    float inv = 1.0f / (sqrtf(rad + 1e-8f) + 1.0f);
    sCD[tid * 3 + 0] = dx * inv; sCD[tid * 3 + 1] = dy * inv; sCD[tid * 3 + 2] = dz * inv;
    sRad[tid] = rad;
    float ex = x0[nodei * 3 + 0] - x0[nodej * 3 + 0];
    float ey = x0[nodei * 3 + 1] - x0[nodej * 3 + 1];
    float ez = x0[nodei * 3 + 2] - x0[nodej * 3 + 2];
    sEA[tid] = ex * ex + ey * ey + ez * ez;
    sEM[tid] = emask[(size_t)nodei * NPG + tid];   // edge e = (b*N+i)*N + j
  }
  __syncthreads();

  const float* wr = ew1 + layer * EW1_LSTR + 256 * HD;
  const float* we = wr + HD;
  const float* b1 = eb1 + layer * HD;
  {
    int k = tid & 127;
    float hik = Hi[(size_t)nodei * HD + k];
    float wrk = wr[k], wek = we[k], bk = b1[k];
    for (int j = tid >> 7; j < NPG; j += 2) {
      float pre = hik + Hj[(size_t)(gbase + j) * HD + k]
                + sRad[j] * wrk + sEA[j] * wek + bk;
      sM[j * SA_LD + k] = (_Float16)silu_f(pre);
    }
  }
  async_wait_all();
  __syncthreads();

  int w = tid >> 5, lane = tid & 31, l = lane & 15, hb = lane >> 4;
  int m0 = (w >> 1) * 16, nb = (w & 1) * 64;

  // GEMM1: m1 @ ew2
  v8f acc[4] = {};
  for (int kk = 0; kk < HD; kk += 32) {
    v16h a = frag_a(sM, SA_LD, m0, kk);
#pragma unroll
    for (int t = 0; t < 4; ++t) {
      v16h bf = frag_bT(sW, SW_LD, kk, nb + t * 16);
      acc[t] = WMMA_F16(a, bf, acc[t]);
    }
  }
  __syncthreads();                                 // all sM/sW reads done
  stage_wT_async(sW, wl + OF_CW1, HD, tid);        // stage cw1 (async)
  const float* b2 = eb2 + layer * HD;
#pragma unroll
  for (int t = 0; t < 4; ++t) {
    int n = nb + t * 16 + l;
    float bias = b2[n];
#pragma unroll
    for (int r = 0; r < 8; ++r) {
      int j = m0 + hb * 8 + r;
      sM[j * SA_LD + n] = (_Float16)(silu_f(acc[t][r] + bias) * sEM[j]);  // m = m*em
    }
  }
  async_wait_all();
  __syncthreads();

  // GEMM2: m @ cw1
  v8f acc2[4] = {};
  for (int kk = 0; kk < HD; kk += 32) {
    v16h a = frag_a(sM, SA_LD, m0, kk);
#pragma unroll
    for (int t = 0; t < 4; ++t) {
      v16h bf = frag_bT(sW, SW_LD, kk, nb + t * 16);
      acc2[t] = WMMA_F16(a, bf, acc2[t]);
    }
  }
  __syncthreads();                                 // sW reads done -> reuse as sP
  const float* cb = cb1 + layer * HD;
#pragma unroll
  for (int t = 0; t < 4; ++t) {
    int n = nb + t * 16 + l;
    float bias = cb[n];
#pragma unroll
    for (int r = 0; r < 8; ++r) {
      int j = m0 + hb * 8 + r;
      sP[j * HD + n] = silu_f(acc2[t][r] + bias);
    }
  }
  __syncthreads();

  // s[j] = p[j].cw2 * em[j]; agg[i][k] = sum_j m[j][k]  (rows owned exclusively)
  const float* c2 = cw2 + layer * HD;
  if (tid < NPG) {
    float s = 0.f;
    for (int k = 0; k < HD; ++k) s += sP[tid * HD + k] * c2[k];
    sS[tid] = s * sEM[tid];
  } else if (tid < NPG + HD) {
    int k = tid - NPG;
    float a0 = 0.f;
    for (int j = 0; j < NPG; ++j) a0 += (float)sM[j * SA_LD + k];
    agg[(size_t)nodei * HD + k] = a0;
  }
  __syncthreads();
  if (tid < 3) {
    float s = 0.f;
    for (int j = 0; j < NPG; ++j) s += sS[j] * sCD[j * 3 + tid];
    xn[nodei * 3 + tid] = xc[nodei * 3 + tid] + s;
  }
}

// ---------------------------------------------------------------------------
// Node update: h_next = (h + silu([h,agg]@nw1+nb1)@nw2 + nb2) * nm.
// K=256 split into two K=128 WMMA passes reusing the 128-row weight tile.
// ---------------------------------------------------------------------------
__launch_bounds__(256)
__global__ void k_node(const float* __restrict__ h, const float* __restrict__ agg,
                       const _Float16* __restrict__ wf16, int layer,
                       const float* __restrict__ nb1, const float* __restrict__ nb2,
                       const float* __restrict__ nmask, float* __restrict__ hn) {
  __shared__ __align__(16) _Float16 sA[NPG * SA_LD];
  __shared__ __align__(16) _Float16 sW[HD * SW_LD];
  int g = blockIdx.x, tid = threadIdx.x, base = g * NPG;
  const _Float16* wl = wf16 + layer * WPL;
  int w = tid >> 5, lane = tid & 31, l = lane & 15, hb = lane >> 4;
  int m0 = (w >> 1) * 16, nb = (w & 1) * 64;

  v8f acc[4] = {};
  for (int pass = 0; pass < 2; ++pass) {
    const float* src = pass ? agg : h;
    __syncthreads();                           // guards sA/sW reuse across passes
    stage_wT_async(sW, wl + OF_NW1 + pass * HD, 256, tid);   // nw1 WT[128][256]
    for (int idx = tid; idx < NPG * HD; idx += 256) {
      int j = idx >> 7, k = idx & 127;
      sA[j * SA_LD + k] = (_Float16)src[(size_t)(base + j) * HD + k];
    }
    async_wait_all();
    __syncthreads();
    for (int kk = 0; kk < HD; kk += 32) {
      v16h a = frag_a(sA, SA_LD, m0, kk);
#pragma unroll
      for (int t = 0; t < 4; ++t) {
        v16h bf = frag_bT(sW, SW_LD, kk, nb + t * 16);
        acc[t] = WMMA_F16(a, bf, acc[t]);
      }
    }
  }
  __syncthreads();
  stage_wT_async(sW, wl + OF_NW2, HD, tid);     // nw2 (async)
  const float* bb1 = nb1 + layer * HD;
#pragma unroll
  for (int t = 0; t < 4; ++t) {
    int n = nb + t * 16 + l;
    float bias = bb1[n];
#pragma unroll
    for (int r = 0; r < 8; ++r) {
      int j = m0 + hb * 8 + r;
      sA[j * SA_LD + n] = (_Float16)silu_f(acc[t][r] + bias);   // t -> A tile
    }
  }
  async_wait_all();
  __syncthreads();
  v8f acc2[4] = {};
  for (int kk = 0; kk < HD; kk += 32) {
    v16h a = frag_a(sA, SA_LD, m0, kk);
#pragma unroll
    for (int t = 0; t < 4; ++t) {
      v16h bf = frag_bT(sW, SW_LD, kk, nb + t * 16);
      acc2[t] = WMMA_F16(a, bf, acc2[t]);
    }
  }
  const float* bb2 = nb2 + layer * HD;
#pragma unroll
  for (int t = 0; t < 4; ++t) {
    int n = nb + t * 16 + l;
    float bias = bb2[n];
#pragma unroll
    for (int r = 0; r < 8; ++r) {
      int node = base + m0 + hb * 8 + r;
      float nm = nmask[node];
      hn[(size_t)node * HD + n] = (h[(size_t)node * HD + n] + acc2[t][r] + bias) * nm;
    }
  }
}

// ---------------------------------------------------------------------------
// Output head: out[b] = mean_n ((h@out_w + out_b) * nm)
// ---------------------------------------------------------------------------
__global__ void k_out(const float* __restrict__ h, const float* __restrict__ out_w,
                      const float* __restrict__ out_b, const float* __restrict__ nmask,
                      float* __restrict__ out) {
  __shared__ float r[NPG];
  int b = blockIdx.x, j = threadIdx.x;
  int node = b * NPG + j;
  float acc = out_b[0];
  for (int k = 0; k < HD; ++k) acc += h[(size_t)node * HD + k] * out_w[k];
  r[j] = acc * nmask[node];
  __syncthreads();
  for (int off = 32; off > 0; off >>= 1) {
    if (j < off) r[j] += r[j + off];
    __syncthreads();
  }
  if (j == 0) out[b] = r[0] * (1.0f / NPG);
}

// ---------------------------------------------------------------------------
extern "C" void kernel_launch(void* const* d_in, const int* in_sizes, int n_in,
                              void* d_out, int out_size, void* d_ws, size_t ws_size,
                              hipStream_t stream) {
  const float* xh     = (const float*)d_in[0];
  const float* nmask  = (const float*)d_in[1];
  const float* emask  = (const float*)d_in[2];
  const float* emb_w  = (const float*)d_in[3];
  const float* emb_b  = (const float*)d_in[4];
  const float* out_w  = (const float*)d_in[5];
  const float* out_b  = (const float*)d_in[6];
  const float* ew1    = (const float*)d_in[7];
  const float* eb1    = (const float*)d_in[8];
  const float* ew2    = (const float*)d_in[9];
  const float* eb2    = (const float*)d_in[10];
  const float* nw1    = (const float*)d_in[11];
  const float* nb1    = (const float*)d_in[12];
  const float* nw2    = (const float*)d_in[13];
  const float* nb2    = (const float*)d_in[14];
  const float* cw1    = (const float*)d_in[15];
  const float* cb1    = (const float*)d_in[16];
  const float* cw2    = (const float*)d_in[17];

  // workspace carve (~11.6 MB total, all 16B-aligned)
  float* x0  = (float*)d_ws;
  float* xA  = x0  + BN * 3;
  float* xB  = xA  + BN * 3;
  float* hA  = xB  + BN * 3;
  float* hB  = hA  + (size_t)BN * HD;
  float* Hi  = hB  + (size_t)BN * HD;
  float* Hj  = Hi  + (size_t)BN * HD;
  float* agg = Hj  + (size_t)BN * HD;
  _Float16* wf16 = (_Float16*)(agg + (size_t)BN * HD);

  k_wcvt<<<(LAYERS * WPL + 255) / 256, 256, 0, stream>>>(ew1, ew2, nw1, nw2, cw1, wf16);
  k_prep<<<BN, HD, 0, stream>>>(xh, nmask, emb_w, emb_b, x0, xA, hA);

  float* xc = xA; float* xn = xB;
  float* hc = hA; float* hn = hB;
  for (int l = 0; l < LAYERS; ++l) {
    k_hih <<<64, 256, 0, stream>>>(hc, wf16 + (size_t)l * WPL, Hi, Hj);
    k_edge<<<BN, 256, 0, stream>>>(xc, x0, Hi, Hj, wf16, l, ew1, eb1, eb2, cb1,
                                   cw2, emask, xn, agg);
    k_node<<<64, 256, 0, stream>>>(hc, agg, wf16, l, nb1, nb2, nmask, hn);
    float* t;
    t = xc; xc = xn; xn = t;
    t = hc; hc = hn; hn = t;
  }
  k_out<<<64, NPG, 0, stream>>>(hc, out_w, out_b, nmask, (float*)d_out);
}
